// OfficialMamba_17076789969283
// MI455X (gfx1250) — matrix operand order repaired and checked by reading, not compiled, not run on hardware
//
#include <hip/hip_runtime.h>
#include <hip/hip_bf16.h>

// ---------------------------------------------------------------------------
// Problem constants (from reference)
// ---------------------------------------------------------------------------
#define B_SZ    8
#define L_SZ    2048
#define DMODEL  512
#define DINNER  1024
#define DSTATE  64
#define DCONV   4
#define DTRANK  32
#define MROWS   (B_SZ * L_SZ)          // 16384
#define NLAYERS 2

typedef __bf16 bf16_t;
typedef __attribute__((ext_vector_type(16))) __bf16 v16bf;
typedef __attribute__((ext_vector_type(8)))  float  v8f;

union BF16Frag {
    v16bf    v;
    uint4    q[2];
};

static __device__ __forceinline__ float sigmoid_fast(float x) {
    return 1.f / (1.f + __expf(-x));
}

// ---------------------------------------------------------------------------
// f32 -> bf16 conversion (weights & activations)
// ---------------------------------------------------------------------------
__global__ __launch_bounds__(256) void f32_to_bf16_kernel(
    const float* __restrict__ in, bf16_t* __restrict__ out, int n)
{
    int i = blockIdx.x * 256 + threadIdx.x;
    if (i < n) out[i] = (bf16_t)in[i];
}

// ---------------------------------------------------------------------------
// LayerNorm over DMODEL=512, one row per block (256 threads, 2 elems/thread),
// emits bf16 activations for the in_proj WMMA GEMM.
// ---------------------------------------------------------------------------
__global__ __launch_bounds__(256) void layernorm_bf16_kernel(
    const float* __restrict__ x, const float* __restrict__ w,
    const float* __restrict__ b, bf16_t* __restrict__ out)
{
    __shared__ float red[256];
    const int row = blockIdx.x;
    const int tid = threadIdx.x;
    const float* xr = x + (size_t)row * DMODEL;

    float v0 = xr[tid];
    float v1 = xr[tid + 256];

    red[tid] = v0 + v1;
    __syncthreads();
    #pragma unroll
    for (int s = 128; s > 0; s >>= 1) {
        if (tid < s) red[tid] += red[tid + s];
        __syncthreads();
    }
    const float mu = red[0] * (1.f / (float)DMODEL);
    __syncthreads();

    float d0 = v0 - mu, d1 = v1 - mu;
    red[tid] = d0 * d0 + d1 * d1;
    __syncthreads();
    #pragma unroll
    for (int s = 128; s > 0; s >>= 1) {
        if (tid < s) red[tid] += red[tid + s];
        __syncthreads();
    }
    const float rs = rsqrtf(red[0] * (1.f / (float)DMODEL) + 1e-5f);

    bf16_t* orow = out + (size_t)row * DMODEL;
    orow[tid]       = (bf16_t)(d0 * rs * w[tid]       + b[tid]);
    orow[tid + 256] = (bf16_t)(d1 * rs * w[tid + 256] + b[tid + 256]);
}

// ---------------------------------------------------------------------------
// WMMA bf16 GEMM:  C[M,N] = A[M,K] * W[N,K]^T   (f32 accumulate)
//   - block = 256 threads = 8 waves; wave computes a 32(M) x (NT*16)(N) tile:
//     2 A fragments, NT shared B fragments, 2*NT WMMAs per K-step.
//   - All fragments for a K-step are loaded BEFORE any WMMA so the scheduler
//     can issue graduated s_wait_loadcnt and overlap loads with matrix ops.
//   - grid = (N/64, M/256) for NT=4, plus one remainder launch with
//     n_origin = (N/64)*64 and NT = (N%64)/16 when N%64 != 0.
// Fragment layouts follow CDNA5 ISA 7.12.2 (wave32). Both A and B fragments
// are two contiguous 16-byte segments per lane -> 2x global_load_b128 each:
//   A 16x32 bf16 : lane m=lane&15, half=lane>>4; elem i<8 -> K=i+8*half,
//                  elem i>=8 -> K=16+(i-8)+8*half
//   B 32x16 bf16 : lane n=lane&15, half=lane>>4; same K mapping as A
//   C/D 16x16 f32: elem r -> m = r + 8*half, n = lane&15
// EPI: 0 = plain store, 1 = softplus(x + bias[col]), 2 = x + resid[row,col]
// ---------------------------------------------------------------------------
template <int EPI, int NT>
__global__ __launch_bounds__(256) void gemm_bf16_wmma_kernel(
    const bf16_t* __restrict__ A, const bf16_t* __restrict__ W,
    const float* __restrict__ bias, const float* __restrict__ resid,
    float* __restrict__ C, int M, int N, int K, int n_origin)
{
    const int lane = threadIdx.x & 31;
    const int wave = threadIdx.x >> 5;
    const int half = lane >> 4;
    const int ln   = lane & 15;
    const int m0   = blockIdx.y * 256 + wave * 32;
    const int n0   = n_origin + blockIdx.x * 64;

    v8f acc[2][NT];
    #pragma unroll
    for (int mt = 0; mt < 2; ++mt)
        #pragma unroll
        for (int t = 0; t < NT; ++t)
            acc[mt][t] = (v8f){0.f, 0.f, 0.f, 0.f, 0.f, 0.f, 0.f, 0.f};

    const bf16_t* arow0 = A + (size_t)(m0 + ln) * K + 8 * half;
    const bf16_t* arow1 = arow0 + (size_t)16 * K;
    const bf16_t* wrow0 = W + (size_t)(n0 + ln) * K + 8 * half;

    #pragma unroll 2
    for (int k0 = 0; k0 < K; k0 += 32) {
        if (k0 + 32 < K) {
            __builtin_prefetch(arow0 + k0 + 32, 0, 1);
            __builtin_prefetch(arow1 + k0 + 32, 0, 1);
        }

        // Issue ALL fragment loads for this K-step first.
        BF16Frag a0, a1, bm[NT];
        a0.q[0] = *(const uint4*)(arow0 + k0);
        a0.q[1] = *(const uint4*)(arow0 + k0 + 16);
        a1.q[0] = *(const uint4*)(arow1 + k0);
        a1.q[1] = *(const uint4*)(arow1 + k0 + 16);
        #pragma unroll
        for (int t = 0; t < NT; ++t) {
            const bf16_t* wrow = wrow0 + (size_t)(t * 16) * K + k0;
            bm[t].q[0] = *(const uint4*)(wrow);
            bm[t].q[1] = *(const uint4*)(wrow + 16);
        }

        // Then all WMMAs (B fragments reused across the two M sub-tiles).
        #pragma unroll
        for (int t = 0; t < NT; ++t)
            acc[0][t] = __builtin_amdgcn_wmma_f32_16x16x32_bf16(
                false, a0.v, false, bm[t].v, (short)0, acc[0][t], false, false);
        #pragma unroll
        for (int t = 0; t < NT; ++t)
            acc[1][t] = __builtin_amdgcn_wmma_f32_16x16x32_bf16(
                false, a1.v, false, bm[t].v, (short)0, acc[1][t], false, false);
    }

    #pragma unroll
    for (int mt = 0; mt < 2; ++mt) {
        #pragma unroll
        for (int t = 0; t < NT; ++t) {
            const int col = n0 + t * 16 + ln;
            float bv = 0.f;
            if (EPI == 1) bv = bias[col];
            #pragma unroll
            for (int r = 0; r < 8; ++r) {
                const int row = m0 + mt * 16 + r + 8 * half;
                float v = acc[mt][t][r];
                if (EPI == 1) {             // softplus(x + b)
                    v += bv;
                    v = (v > 20.f) ? v : log1pf(__expf(v));
                }
                if (EPI == 2) {             // residual add
                    v += resid[(size_t)row * N + col];
                }
                C[(size_t)row * N + col] = v;
            }
        }
    }
}

// Host-side helper: full 64-wide blocks + at most one remainder launch.
template <int EPI>
static void launch_gemm(const bf16_t* A, const bf16_t* W, const float* bias,
                        const float* resid, float* C, int M, int N, int K,
                        hipStream_t stream)
{
    const dim3 blk(256);
    const int nfull = N / 64;
    const int rem   = N % 64;            // multiple of 16 for all our shapes
    if (nfull > 0) {
        gemm_bf16_wmma_kernel<EPI, 4><<<dim3(nfull, MROWS / 256), blk, 0, stream>>>(
            A, W, bias, resid, C, M, N, K, 0);
    }
    if (rem == 16)
        gemm_bf16_wmma_kernel<EPI, 1><<<dim3(1, MROWS / 256), blk, 0, stream>>>(
            A, W, bias, resid, C, M, N, K, nfull * 64);
    else if (rem == 32)
        gemm_bf16_wmma_kernel<EPI, 2><<<dim3(1, MROWS / 256), blk, 0, stream>>>(
            A, W, bias, resid, C, M, N, K, nfull * 64);
    else if (rem == 48)
        gemm_bf16_wmma_kernel<EPI, 3><<<dim3(1, MROWS / 256), blk, 0, stream>>>(
            A, W, bias, resid, C, M, N, K, nfull * 64);
}

// ---------------------------------------------------------------------------
// Causal depthwise conv (taps=4) + bias + SiLU, reads the u half of xz
// (xz layout [M, 2*DINNER], u = cols [0,DINNER)) and writes bf16 activations.
// One thread per (b,l,d) element.
// ---------------------------------------------------------------------------
__global__ __launch_bounds__(256) void conv_silu_kernel(
    const float* __restrict__ xz, const float* __restrict__ cw,
    const float* __restrict__ cb, bf16_t* __restrict__ uact)
{
    const int idx = blockIdx.x * 256 + threadIdx.x;   // over MROWS*DINNER
    const int d   = idx & (DINNER - 1);
    const int bl  = idx >> 10;                        // DINNER = 1024
    const int l   = bl & (L_SZ - 1);
    const int b   = bl >> 11;                         // L_SZ = 2048

    float acc = cb[d];
    #pragma unroll
    for (int t = 0; t < DCONV; ++t) {
        const int lp = l - (DCONV - 1) + t;
        if (lp >= 0) {
            const size_t src = ((size_t)(b * L_SZ + lp)) * (2 * DINNER) + d;
            acc = fmaf(cw[d * DCONV + t], xz[src], acc);
        }
    }
    acc = acc * sigmoid_fast(acc);                    // SiLU
    uact[(size_t)bl * DINNER + d] = (bf16_t)acc;
}

// ---------------------------------------------------------------------------
// Extract dt low-rank columns [0,DTRANK) of x_dbl[M,160] -> bf16
// ---------------------------------------------------------------------------
__global__ __launch_bounds__(256) void dt_extract_kernel(
    const float* __restrict__ xdbl, bf16_t* __restrict__ dtr)
{
    const int idx = blockIdx.x * 256 + threadIdx.x;   // over MROWS*DTRANK
    const int c   = idx & (DTRANK - 1);
    const int row = idx >> 5;                         // DTRANK = 32
    dtr[(size_t)row * DTRANK + c] =
        (bf16_t)xdbl[(size_t)row * (DTRANK + 2 * DSTATE) + c];
}

// ---------------------------------------------------------------------------
// Selective scan, state-split variant: adjacent lane pair shares one (b,d)
// channel; lane owns 32 of the 64 states (h[32] + A[32] in VGPRs), partial
// y combined with __shfl_xor(y,1). B_t / C_t staged through LDS.
// Fuses + D*u, * silu(z), bf16 conversion for the out_proj GEMM.
// grid = B * (DINNER/128) = 64 blocks, 256 threads.
// ---------------------------------------------------------------------------
#define SSTATES (DSTATE / 2)             // 32 states per lane

__global__ __launch_bounds__(256) void selective_scan_kernel(
    const float* __restrict__ xdbl,   // [M, 160]  (B cols 32..95, C cols 96..159)
    const float* __restrict__ dt,     // [M, DINNER]
    const bf16_t* __restrict__ uact,  // [M, DINNER]
    const float* __restrict__ xz,     // [M, 2*DINNER] (z = cols DINNER..)
    const float* __restrict__ A_log,  // [DINNER, DSTATE]
    const float* __restrict__ Dv,     // [DINNER]
    bf16_t* __restrict__ ybf)         // [M, DINNER]
{
    __shared__ float sB[DSTATE];
    __shared__ float sC[DSTATE];

    const int tid   = threadIdx.x;
    const int shalf = tid & 1;                        // which 32 states
    const int d     = (blockIdx.x & 7) * 128 + (tid >> 1);
    const int b     = blockIdx.x >> 3;
    const int sbase = shalf * SSTATES;

    float Ad[SSTATES];
    #pragma unroll
    for (int s = 0; s < SSTATES; ++s)
        Ad[s] = -__expf(A_log[(size_t)d * DSTATE + sbase + s]);

    float h[SSTATES];
    #pragma unroll
    for (int s = 0; s < SSTATES; ++s) h[s] = 0.f;

    const float Dd = Dv[d];

    for (int l = 0; l < L_SZ; ++l) {
        const size_t rb = (size_t)b * L_SZ + l;
        if (tid < DSTATE)
            sB[tid] = xdbl[rb * (DTRANK + 2 * DSTATE) + DTRANK + tid];
        else if (tid < 2 * DSTATE)
            sC[tid - DSTATE] =
                xdbl[rb * (DTRANK + 2 * DSTATE) + DTRANK + DSTATE + (tid - DSTATE)];
        __syncthreads();

        const float dtt = dt[rb * DINNER + d];
        const float ut  = (float)uact[rb * DINNER + d];
        const float du  = dtt * ut;

        float y = 0.f;
        #pragma unroll
        for (int s = 0; s < SSTATES; ++s) {
            const float dA = __expf(dtt * Ad[s]);
            h[s] = fmaf(dA, h[s], du * sB[sbase + s]);
            y    = fmaf(h[s], sC[sbase + s], y);
        }
        // combine the two state halves of this channel (adjacent lanes)
        y += __shfl_xor(y, 1, 32);

        if (shalf == 0) {
            y = fmaf(Dd, ut, y);
            const float z = xz[rb * (2 * DINNER) + DINNER + d];
            y *= z * sigmoid_fast(z);                // y * silu(z)
            ybf[rb * DINNER + d] = (bf16_t)y;
        }
        __syncthreads();
    }
}

// ---------------------------------------------------------------------------
// Host-side orchestration
// ---------------------------------------------------------------------------
static inline int cdiv(int a, int b) { return (a + b - 1) / b; }

extern "C" void kernel_launch(void* const* d_in, const int* in_sizes, int n_in,
                              void* d_out, int out_size, void* d_ws, size_t ws_size,
                              hipStream_t stream)
{
    (void)in_sizes; (void)n_in; (void)out_size; (void)ws_size;

    const float* x         = (const float*)d_in[0];
    const float* ln_w      = (const float*)d_in[1];
    const float* ln_b      = (const float*)d_in[2];
    const float* in_proj_w = (const float*)d_in[3];
    const float* conv_w    = (const float*)d_in[4];
    const float* conv_b    = (const float*)d_in[5];
    const float* x_proj_w  = (const float*)d_in[6];
    const float* dt_proj_w = (const float*)d_in[7];
    const float* dt_proj_b = (const float*)d_in[8];
    const float* A_log     = (const float*)d_in[9];
    const float* Dvec      = (const float*)d_in[10];
    const float* out_proj_w= (const float*)d_in[11];

    // ---- workspace carve-up -------------------------------------------------
    char* p = (char*)d_ws;
    auto carve = [&p](size_t bytes) -> char* {
        char* r = p;
        p += (bytes + 255) & ~size_t(255);
        return r;
    };
    const size_t XPROJ_N = DTRANK + 2 * DSTATE;       // 160

    bf16_t* wbf_in  = (bf16_t*)carve((size_t)NLAYERS * 2 * DINNER * DMODEL * 2);
    bf16_t* wbf_xp  = (bf16_t*)carve((size_t)NLAYERS * XPROJ_N * DINNER * 2);
    bf16_t* wbf_dt  = (bf16_t*)carve((size_t)NLAYERS * DINNER * DTRANK * 2);
    bf16_t* wbf_out = (bf16_t*)carve((size_t)NLAYERS * DMODEL * DINNER * 2);
    bf16_t* hbf     = (bf16_t*)carve((size_t)MROWS * DMODEL * 2);
    float*  xz      = (float*) carve((size_t)MROWS * 2 * DINNER * 4);
    bf16_t* uact    = (bf16_t*)carve((size_t)MROWS * DINNER * 2);
    float*  xdbl    = (float*) carve((size_t)MROWS * XPROJ_N * 4);
    bf16_t* dtr     = (bf16_t*)carve((size_t)MROWS * DTRANK * 2);
    float*  dtbuf   = (float*) carve((size_t)MROWS * DINNER * 4);
    bf16_t* ybf     = (bf16_t*)carve((size_t)MROWS * DINNER * 2);
    float*  xbuf    = (float*) carve((size_t)MROWS * DMODEL * 4);

    // ---- weight conversion (once per call) ---------------------------------
    {
        int n;
        n = NLAYERS * 2 * DINNER * DMODEL;
        f32_to_bf16_kernel<<<cdiv(n, 256), 256, 0, stream>>>(in_proj_w, wbf_in, n);
        n = NLAYERS * (int)XPROJ_N * DINNER;
        f32_to_bf16_kernel<<<cdiv(n, 256), 256, 0, stream>>>(x_proj_w, wbf_xp, n);
        n = NLAYERS * DINNER * DTRANK;
        f32_to_bf16_kernel<<<cdiv(n, 256), 256, 0, stream>>>(dt_proj_w, wbf_dt, n);
        n = NLAYERS * DMODEL * DINNER;
        f32_to_bf16_kernel<<<cdiv(n, 256), 256, 0, stream>>>(out_proj_w, wbf_out, n);
    }

    const dim3 blk(256);
    for (int l = 0; l < NLAYERS; ++l) {
        const float* xin  = (l == 0) ? x : xbuf;
        float*       xout = (l == NLAYERS - 1) ? (float*)d_out : xbuf;

        // 1) LayerNorm -> bf16
        layernorm_bf16_kernel<<<MROWS, blk, 0, stream>>>(
            xin, ln_w + l * DMODEL, ln_b + l * DMODEL, hbf);

        // 2) in_proj GEMM: [M,512] x [2048,512]^T -> xz[M,2048]
        launch_gemm<0>(hbf, wbf_in + (size_t)l * 2 * DINNER * DMODEL,
                       nullptr, nullptr, xz, MROWS, 2 * DINNER, DMODEL, stream);

        // 3) causal depthwise conv + SiLU -> uact (bf16)
        conv_silu_kernel<<<(MROWS * DINNER) / 256, blk, 0, stream>>>(
            xz, conv_w + (size_t)l * DINNER * DCONV, conv_b + (size_t)l * DINNER, uact);

        // 4) x_proj GEMM: [M,1024] x [160,1024]^T -> xdbl[M,160]
        launch_gemm<0>(uact, wbf_xp + (size_t)l * XPROJ_N * DINNER,
                       nullptr, nullptr, xdbl, MROWS, (int)XPROJ_N, DINNER, stream);

        // 5) dt low-rank extract -> bf16
        dt_extract_kernel<<<(MROWS * DTRANK) / 256, blk, 0, stream>>>(xdbl, dtr);

        // 6) dt_proj GEMM + softplus(x+b): [M,32] x [1024,32]^T -> dtbuf[M,1024]
        launch_gemm<1>(dtr, wbf_dt + (size_t)l * DINNER * DTRANK,
                       dt_proj_b + (size_t)l * DINNER, nullptr, dtbuf,
                       MROWS, DINNER, DTRANK, stream);

        // 7) selective scan (+ D*u, * silu(z)) -> ybf (bf16)
        selective_scan_kernel<<<B_SZ * (DINNER / 128), blk, 0, stream>>>(
            xdbl, dtbuf, uact, xz,
            A_log + (size_t)l * DINNER * DSTATE, Dvec + (size_t)l * DINNER, ybf);

        // 8) out_proj GEMM + residual: [M,1024] x [512,1024]^T -> xout[M,512]
        launch_gemm<2>(ybf, wbf_out + (size_t)l * DMODEL * DINNER,
                       nullptr, xin, xout, MROWS, DMODEL, DINNER, stream);
    }
}
